// SocialAttention_49890340110384
// MI455X (gfx1250) — matrix operand order repaired and checked by reading, not compiled, not run on hardware
//
#include <hip/hip_runtime.h>

typedef __attribute__((ext_vector_type(16))) _Float16     v16h;
typedef __attribute__((ext_vector_type(8)))  float        v8f;
typedef __attribute__((ext_vector_type(8)))  unsigned int uv8;
typedef __attribute__((ext_vector_type(4)))  unsigned int u32x4;
typedef __attribute__((ext_vector_type(2)))  unsigned int u32x2;

#define IN_DIM 64
#define HID    128
#define NHUM   256
#define BATCH  4096

union Frag { u32x4 q[2]; uv8 u; v16h h; };
union PK   { unsigned int u; _Float16 h[2]; };

// ---- A-fragment (weight^T tile), prepacked [tile][lane][8 dwords] ----------
__device__ __forceinline__ v16h load_w_frag(const u32x4* __restrict__ wsA4,
                                            int tile, int lane) {
  Frag f;
  const u32x4* p = wsA4 + tile * 64 + lane * 2;   // 32B contiguous per lane
  f.q[0] = p[0];                                  // global_load_b128
  f.q[1] = p[1];                                  // global_load_b128
  return f.h;
}

// ---- B-fragment of X^T where X is row-major [row][feat] f16 in LDS ---------
// B layout (32x16): lane = column (= data row m), lanes 0-15 hold K 0-15,
// lanes 16-31 hold K 16-31, 2 K per dword -> 8 consecutive dwords of row m.
__device__ __forceinline__ v16h load_bT_frag(const u32x4* a4, int rowHalves,
                                             int mbase, int kt, int lane) {
  Frag f;
  const int m  = mbase + (lane & 15);
  const int i4 = (m * rowHalves + kt * 32 + ((lane >> 4) << 4)) >> 3;
  f.q[0] = a4[i4];        // ds_load_b128
  f.q[1] = a4[i4 + 1];    // ds_load_b128
  return f.h;
}

// ---------------------------------------------------------------------------
// Kernel 0: pack W_h^T (A-frags over d, rows=h) and W1[H:]^T (A-frags over j,
// rows=o). A layout: lane = row (L&15); K 0-15 (lanes 0-15) /
// 16-31 (lanes 16-31), 2 K per dword.
__global__ void pack_weights(const float* __restrict__ W_h,
                             const float* __restrict__ W1,
                             unsigned int* __restrict__ WhT,
                             unsigned int* __restrict__ W1T) {
  const int t = blockIdx.x * blockDim.x + threadIdx.x; // 0..12287
  if (t >= 12288) return;
  const bool isW1 = (t >= 4096);
  const int idx  = isW1 ? (t - 4096) : t;
  const int lane = idx & 31;
  const int i    = (idx >> 5) & 7;
  const int tile = idx >> 8;                  // kt*8 + rt
  const int rt   = tile & 7;
  const int kt   = tile >> 3;
  const int rr   = rt * 16 + (lane & 15);     // output row (h or o)
  const int k    = kt * 32 + ((lane >> 4) << 4) + (i << 1);
  const int dst  = tile * 256 + lane * 8 + i; // [tile][lane][i]
  PK pk;
  if (!isW1) {               // W_h^T[rr][k] = W_h[k][rr],  W_h is (64,128)
    pk.h[0] = (_Float16)W_h[k * HID + rr];
    pk.h[1] = (_Float16)W_h[(k + 1) * HID + rr];
    WhT[dst] = pk.u;
  } else {                   // W1p^T[rr][k] = W1[H+k][rr], W1 is (256,128)
    const int kg = HID + k;
    pk.h[0] = (_Float16)W1[kg * HID + rr];
    pk.h[1] = (_Float16)W1[(kg + 1) * HID + rr];
    W1T[dst] = pk.u;
  }
}

// ---------------------------------------------------------------------------
// Kernel 1: r1b[b] = relu(robot_state[b] @ W_r + b_r) @ W1[:H]   (tiny)
__global__ __launch_bounds__(128) void robot_prep(
    const float* __restrict__ robot_state, const float* __restrict__ W_r,
    const float* __restrict__ b_r, const float* __restrict__ W1,
    float* __restrict__ r1b) {
  __shared__ float emb[HID];
  __shared__ float rs[IN_DIM];
  const int b = blockIdx.x, t = threadIdx.x;
  if (t < IN_DIM) rs[t] = robot_state[b * IN_DIM + t];
  __syncthreads();
  float a = b_r[t];
  for (int i = 0; i < IN_DIM; ++i) a += rs[i] * W_r[i * HID + t];
  emb[t] = fmaxf(a, 0.0f);
  __syncthreads();
  float r = 0.0f;
  for (int h = 0; h < HID; ++h) r += emb[h] * W1[h * HID + t];
  r1b[(size_t)b * HID + t] = r;
}

// ---------------------------------------------------------------------------
// Kernel 2: fused per-batch-row attention, transposed WMMA orientation.
// 8 waves; wave w owns humans 32w..32w+31 (two 16-column tiles).
__global__ __launch_bounds__(256) void social_main(
    const float* __restrict__ human_states, const float* __restrict__ b_h,
    const float* __restrict__ b1, const float* __restrict__ W2,
    const float* __restrict__ b2, const float* __restrict__ W_o,
    const float* __restrict__ b_o, const float* __restrict__ r1b,
    const unsigned int* __restrict__ WhT, const unsigned int* __restrict__ W1T,
    float* __restrict__ out) {
  extern __shared__ char smem[];
  _Float16* hs16   = (_Float16*)smem;                        // 256*64  f16
  _Float16* hemb16 = (_Float16*)(smem + 32768);              // 256*128 f16
  float*    scores = (float*)(smem + 32768 + 65536);         // 256
  float*    red    = scores + NHUM;                          // 16
  float*    attp   = red + 16;                               // 8*64
  float*    att    = attp + 512;                             // 64

  const int b    = blockIdx.x;
  const int tid  = threadIdx.x;
  const int lane = tid & 31;
  const int wave = tid >> 5;
  const int mb0  = wave * 32;

  const u32x4* WhT4 = (const u32x4*)WhT;
  const u32x4* W1T4 = (const u32x4*)W1T;

  // ---- load human_states[b] once, convert to f16, packed b64 LDS stores ----
  const float4* src = (const float4*)(human_states + (size_t)b * NHUM * IN_DIM);
  u32x2* hsq = (u32x2*)hs16;
  for (int i = tid; i < NHUM * IN_DIM / 4; i += 256) {
    float4 v = src[i];
    PK a, c;
    a.h[0] = (_Float16)v.x; a.h[1] = (_Float16)v.y;
    c.h[0] = (_Float16)v.z; c.h[1] = (_Float16)v.w;
    u32x2 d; d.x = a.u; d.y = c.u;
    hsq[i] = d;
  }
  __syncthreads();

  const int mcol  = lane & 15;
  const int hhalf = (lane >> 4) << 3;          // 0 or 8

  // ---- Stage A: hemb^T tiles = W_h^T @ hs^T; store row-major hemb ----------
  {
    const u32x4* hsA = (const u32x4*)hs16;
    // hoist loop-invariant hs^T B-fragments (2 human tiles x 2 k-tiles)
    v16h bf00 = load_bT_frag(hsA, IN_DIM, mb0,      0, lane);
    v16h bf01 = load_bT_frag(hsA, IN_DIM, mb0,      1, lane);
    v16h bf10 = load_bT_frag(hsA, IN_DIM, mb0 + 16, 0, lane);
    v16h bf11 = load_bT_frag(hsA, IN_DIM, mb0 + 16, 1, lane);
    for (int ht = 0; ht < 8; ++ht) {
      v16h a0 = load_w_frag(WhT4, ht,     lane);
      v16h a1 = load_w_frag(WhT4, 8 + ht, lane);
      v8f c0 = {}; v8f c1 = {};
      c0 = __builtin_amdgcn_wmma_f32_16x16x32_f16(false, a0, false, bf00, (short)0, c0, false, false);
      c1 = __builtin_amdgcn_wmma_f32_16x16x32_f16(false, a0, false, bf10, (short)0, c1, false, false);
      c0 = __builtin_amdgcn_wmma_f32_16x16x32_f16(false, a1, false, bf01, (short)0, c0, false, false);
      c1 = __builtin_amdgcn_wmma_f32_16x16x32_f16(false, a1, false, bf11, (short)0, c1, false, false);
      const int h0 = ht * 16 + hhalf;          // 8 consecutive h per lane
      float4 ba = *(const float4*)(b_h + h0);
      float4 bb = *(const float4*)(b_h + h0 + 4);
      const float bias[8] = {ba.x, ba.y, ba.z, ba.w, bb.x, bb.y, bb.z, bb.w};
      u32x4 o0, o1;
#pragma unroll
      for (int j = 0; j < 4; ++j) {
        PK q0, q1;
        q0.h[0] = (_Float16)fmaxf(c0[2 * j]     + bias[2 * j],     0.0f);
        q0.h[1] = (_Float16)fmaxf(c0[2 * j + 1] + bias[2 * j + 1], 0.0f);
        q1.h[0] = (_Float16)fmaxf(c1[2 * j]     + bias[2 * j],     0.0f);
        q1.h[1] = (_Float16)fmaxf(c1[2 * j + 1] + bias[2 * j + 1], 0.0f);
        o0[j] = q0.u; o1[j] = q1.u;
      }
      *(u32x4*)(hemb16 + (mb0 +      mcol) * HID + h0) = o0;  // ds_store_b128
      *(u32x4*)(hemb16 + (mb0 + 16 + mcol) * HID + h0) = o1;  // ds_store_b128
    }
  }
  __syncthreads();

  // ---- Stage B: hid^T = W1p^T @ hemb^T; fold into scores in-lane -----------
  const float* r1  = r1b + (size_t)b * HID;
  const float  b2v = b2[0];
  {
    const u32x4* heA = (const u32x4*)hemb16;
    // hoist loop-invariant hemb^T B-fragments (2 human tiles x 4 k-tiles)
    v16h bg0[4], bg1[4];
#pragma unroll
    for (int kt = 0; kt < 4; ++kt) {
      bg0[kt] = load_bT_frag(heA, HID, mb0,      kt, lane);
      bg1[kt] = load_bT_frag(heA, HID, mb0 + 16, kt, lane);
    }
    float sp0 = 0.0f, sp1 = 0.0f;
    for (int ot = 0; ot < 8; ++ot) {
      v8f c0 = {}; v8f c1 = {};
#pragma unroll
      for (int kt = 0; kt < 4; ++kt) {
        v16h af = load_w_frag(W1T4, kt * 8 + ot, lane);
        c0 = __builtin_amdgcn_wmma_f32_16x16x32_f16(false, af, false, bg0[kt], (short)0, c0, false, false);
        c1 = __builtin_amdgcn_wmma_f32_16x16x32_f16(false, af, false, bg1[kt], (short)0, c1, false, false);
      }
      const int o0i = ot * 16 + hhalf;
      float4 ra = *(const float4*)(r1 + o0i);
      float4 rb = *(const float4*)(r1 + o0i + 4);
      float4 ca = *(const float4*)(b1 + o0i);
      float4 cb = *(const float4*)(b1 + o0i + 4);
      float4 wa = *(const float4*)(W2 + o0i);
      float4 wb = *(const float4*)(W2 + o0i + 4);
      const float add[8] = {ra.x + ca.x, ra.y + ca.y, ra.z + ca.z, ra.w + ca.w,
                            rb.x + cb.x, rb.y + cb.y, rb.z + cb.z, rb.w + cb.w};
      const float w2v[8] = {wa.x, wa.y, wa.z, wa.w, wb.x, wb.y, wb.z, wb.w};
#pragma unroll
      for (int r = 0; r < 8; ++r) {
        sp0 += fmaxf(c0[r] + add[r], 0.0f) * w2v[r];
        sp1 += fmaxf(c1[r] + add[r], 0.0f) * w2v[r];
      }
    }
    // rows o split across lane halves: one cross-half combine per human tile
    sp0 += __shfl_xor(sp0, 16, 32);
    sp1 += __shfl_xor(sp1, 16, 32);
    if (lane < 16) {
      scores[mb0 + lane]      = sp0 + b2v;
      scores[mb0 + 16 + lane] = sp1 + b2v;
    }
  }
  __syncthreads();

  // ---- softmax over 256 scores --------------------------------------------
  const float s = scores[tid];
  float vmax = s;
  for (int sh = 1; sh < 32; sh <<= 1) vmax = fmaxf(vmax, __shfl_xor(vmax, sh, 32));
  if (lane == 0) red[wave] = vmax;
  __syncthreads();
  vmax = red[0];
#pragma unroll
  for (int i = 1; i < 8; ++i) vmax = fmaxf(vmax, red[i]);
  const float e = __expf(s - vmax);
  float vsum = e;
  for (int sh = 1; sh < 32; sh <<= 1) vsum += __shfl_xor(vsum, sh, 32);
  if (lane == 0) red[8 + wave] = vsum;
  __syncthreads();
  vsum = 0.0f;
#pragma unroll
  for (int i = 0; i < 8; ++i) vsum += red[8 + i];
  scores[tid] = e / vsum;                       // reuse as weights
  __syncthreads();

  // ---- attended = weights^T @ hs (f16-pair loads) --------------------------
  {
    const int dp    = (tid & 31) << 1;
    const int chunk = tid >> 5;
    float acc0 = 0.0f, acc1 = 0.0f;
    const unsigned int* hsrow = (const unsigned int*)hs16;
    for (int nn = chunk * 32; nn < chunk * 32 + 32; ++nn) {
      const float w = scores[nn];
      PK pk; pk.u = hsrow[(nn * IN_DIM + dp) >> 1];
      acc0 += w * (float)pk.h[0];
      acc1 += w * (float)pk.h[1];
    }
    attp[chunk * 64 + dp]     = acc0;
    attp[chunk * 64 + dp + 1] = acc1;
  }
  __syncthreads();
  if (tid < 64) {
    float a = 0.0f;
#pragma unroll
    for (int c = 0; c < 8; ++c) a += attp[c * 64 + tid];
    att[tid] = a;
  }
  __syncthreads();
  if (tid < 128) {
    float o = b_o[tid];
#pragma unroll
    for (int dd = 0; dd < 64; ++dd) o += att[dd] * W_o[dd * 128 + tid];
    out[(size_t)b * 128 + tid] = o;
  }
}

// ---------------------------------------------------------------------------
extern "C" void kernel_launch(void* const* d_in, const int* in_sizes, int n_in,
                              void* d_out, int out_size, void* d_ws, size_t ws_size,
                              hipStream_t stream) {
  (void)in_sizes; (void)n_in; (void)out_size; (void)ws_size;
  const float* robot = (const float*)d_in[0];
  const float* human = (const float*)d_in[1];
  const float* W_r   = (const float*)d_in[2];
  const float* b_r   = (const float*)d_in[3];
  const float* W_h   = (const float*)d_in[4];
  const float* b_h   = (const float*)d_in[5];
  const float* W1    = (const float*)d_in[6];
  const float* b1    = (const float*)d_in[7];
  const float* W2    = (const float*)d_in[8];
  const float* b2    = (const float*)d_in[9];
  const float* W_o   = (const float*)d_in[10];
  const float* b_o   = (const float*)d_in[11];
  float* out = (float*)d_out;

  float*        r1b = (float*)d_ws;                                    // 4096*128 f32
  unsigned int* WhT = (unsigned int*)((char*)d_ws + (size_t)BATCH * HID * 4);
  unsigned int* W1T = WhT + 16 * 256;                                  // after 16 tiles

  pack_weights<<<48, 256, 0, stream>>>(W_h, W1, WhT, W1T);
  robot_prep<<<BATCH, 128, 0, stream>>>(robot, W_r, b_r, W1, r1b);

  const size_t smem = 32768 + 65536 + (size_t)(NHUM + 16 + 512 + 64) * sizeof(float);
  social_main<<<BATCH, 256, smem, stream>>>(human, b_h, b1, W2, b2, W_o, b_o,
                                            r1b, WhT, W1T, out);
}
